// QuantumGNN_24438363914909
// MI455X (gfx1250) — matrix operand order repaired and checked by reading, not compiled, not run on hardware
//
#include <hip/hip_runtime.h>
#include <hip/hip_bf16.h>
#include <cmath>

typedef __attribute__((ext_vector_type(16))) _Float16 v16h;
typedef __attribute__((ext_vector_type(8)))  _Float16 v8h;
typedef __attribute__((ext_vector_type(8)))  float    v8f;

#define N_NODES 512
#define EDGES   16384
#define NODE_DIM 64
#define HID     128
#define FFD     2048
#define VOCAB   50
#define LAYERS  2
#define HEADS   8
#define SEQ     100
#define SP      112      // seq padded to 7*16
#define KP      128      // t-dim padded to 4*32 for P@V
#define MTOK    (SEQ*N_NODES)   // 51200 tokens

union AF { v16h v; v8h h[2]; };

__device__ __forceinline__ v8f wmma_f16(v16h a, v16h b, v8f c) {
  return __builtin_amdgcn_wmma_f32_16x16x32_f16(false, a, false, b, (short)0, c, false, false);
}

// ---- CDNA5 async global->LDS copy (16B per lane), ASYNCcnt tracked --------
__device__ __forceinline__ void async_ld16(const void* g, void* lds) {
  unsigned off = (unsigned)(size_t)lds;   // low 32 bits of generic LDS ptr = LDS byte addr
  asm volatile("global_load_async_to_lds_b128 %0, %1, off"
               :: "v"(off), "v"(g) : "memory");
}
__device__ __forceinline__ void async_wait0() {
#if __has_builtin(__builtin_amdgcn_s_wait_asynccnt)
  __builtin_amdgcn_s_wait_asynccnt(0);
#else
  asm volatile("s_wait_asynccnt 0" ::: "memory");
#endif
}

// ---------------------------------------------------------------- utilities
__global__ void k_zero_f32(float* p, size_t n) {
  size_t i = (size_t)blockIdx.x * blockDim.x + threadIdx.x;
  if (i < n) p[i] = 0.f;
}
__global__ void k_zero_f16(_Float16* p, size_t n) {
  size_t i = (size_t)blockIdx.x * blockDim.x + threadIdx.x;
  if (i < n) p[i] = (_Float16)0.f;
}

// W [K x N] row-major f32  ->  dst [Nd x K] f16 (transposed, zero-padded rows)
__global__ void k_convT(const float* __restrict__ W, _Float16* __restrict__ dst,
                        int K, int N, int Nd) {
  int i = blockIdx.x * blockDim.x + threadIdx.x;
  if (i >= Nd * K) return;
  int n = i / K, k = i - n * K;
  float v = (n < N) ? W[(size_t)k * N + n] : 0.f;
  dst[i] = (_Float16)v;
}

// quantum embedding: qx = (cos(.05)-sin(.05)) * (cos(.1)*x + sin(.1)*tanh(x))
__global__ void k_quantum(const float* __restrict__ x, _Float16* __restrict__ q,
                          float c1, float c2, float c3, int n) {
  int i = blockIdx.x * blockDim.x + threadIdx.x;
  if (i < n) {
    float v = x[i];
    q[i] = (_Float16)(c1 * (c2 * v + c3 * tanhf(v)));
  }
}

// ---------------------------------------------------------------- GCN pieces
__global__ void k_deg_init(float* deg) {
  int i = blockIdx.x * blockDim.x + threadIdx.x;
  if (i < N_NODES) deg[i] = 1.f;   // self loop
}
__global__ void k_deg_count(const int* __restrict__ dst, float* deg) {
  int e = blockIdx.x * blockDim.x + threadIdx.x;
  if (e < EDGES) atomicAdd(&deg[dst[e]], 1.f);
}
__global__ void k_deg_rsqrt(float* deg) {
  int i = blockIdx.x * blockDim.x + threadIdx.x;
  if (i < N_NODES) deg[i] = rsqrtf(deg[i]);   // now holds dinv
}
__global__ void k_edge_agg(const int* __restrict__ src, const int* __restrict__ dst,
                           const float* __restrict__ dinv, const float* __restrict__ lin,
                           float* __restrict__ agg) {
  int e = blockIdx.x;          // one block per edge
  int c = threadIdx.x;         // 128 channels
  int s = src[e], d = dst[e];
  float w = dinv[s] * dinv[d];
  atomicAdd(&agg[(size_t)d * HID + c], lin[(size_t)s * HID + c] * w);
}
__global__ void k_gcn_post(const float* __restrict__ lin, const float* __restrict__ dinv,
                           const float* __restrict__ agg, const float* __restrict__ bias,
                           float* __restrict__ hF, _Float16* __restrict__ hH) {
  int i = blockIdx.x * blockDim.x + threadIdx.x;
  if (i >= N_NODES * HID) return;
  int n = i >> 7, c = i & 127;
  float v = agg[i] + lin[i] * dinv[n] * dinv[n] + bias[c];  // self-loop term + bias
  v = fmaxf(v, 0.f);
  hF[i] = v; hH[i] = (_Float16)v;
}

// ---------------------------------------------------------------- WMMA GEMM
// C[M x 128] = A[M x K] f16 * Bt^T (Bt stored [N x K] f16) + bias (+resid)(relu)
// B tile (16 x K, shared by all 8 waves) is staged once per block into LDS
// via GLOBAL_LOAD_ASYNC_TO_LDS_B128.
// storeMode 0: f32 out (ld=HID) + optional f16 mirror
// storeMode 1: pack f16 into q/k layout [n][h][SP][16]
// storeMode 2: pack f16 into vT layout  [n][h][16][KP]
// storeMode 3: f32 out ld=VOCAB, cols < VOCAB only
__global__ void k_gemm(const _Float16* __restrict__ A, const _Float16* __restrict__ Bt,
                       const float* __restrict__ bias, float* __restrict__ outF,
                       _Float16* __restrict__ outH, const float* __restrict__ resid,
                       int M, int K, int flags, int storeMode) {
  __shared__ _Float16 btile[16 * 128];          // max K = 128
  int tid = threadIdx.x;
  int wave = tid >> 5, lane = tid & 31;
  int mr = lane & 15, hi = lane >> 4;
  int m0 = blockIdx.x * 128 + wave * 16;
  int n0 = blockIdx.y * 16;

  // stage B tile: 16*K halves, 8 halves (16B) per lane, async DMA to LDS
  if (tid < 2 * K)
    async_ld16(Bt + (size_t)n0 * K + tid * 8, btile + tid * 8);
  async_wait0();
  __syncthreads();

  if (m0 < M) {
    int col = n0 + mr;
    const _Float16* arow = A + (size_t)(m0 + mr) * K;
    const _Float16* brow = btile + (size_t)mr * K;
    v8f c = {};
    for (int k0 = 0; k0 < K; k0 += 32) {
      AF a;
      a.v = *(const v16h*)(arow + k0 + hi * 16);
      v16h b = *(const v16h*)(brow + k0 + hi * 16);
      c = wmma_f16(a.v, b, c);
    }
    float bv = 0.f;
    if (bias && (storeMode != 3 || col < VOCAB)) bv = bias[col];
#pragma unroll
    for (int i = 0; i < 8; i++) {
      size_t row = (size_t)m0 + hi * 8 + i;
      float v = c[i] + bv;
      if (flags & 2) v += resid[row * HID + col];
      if (flags & 1) v = fmaxf(v, 0.f);
      if (storeMode == 0) {
        outF[row * HID + col] = v;
        if (outH) outH[row * HID + col] = (_Float16)v;
      } else if (storeMode == 1) {
        int s = (int)(row >> 9), n = (int)(row & 511);
        outH[(((size_t)(n * HEADS + (col >> 4))) * SP + s) * 16 + (col & 15)] = (_Float16)v;
      } else if (storeMode == 2) {
        int s = (int)(row >> 9), n = (int)(row & 511);
        outH[(((size_t)(n * HEADS + (col >> 4))) * 16 + (col & 15)) * KP + s] = (_Float16)v;
      } else {
        if (col < VOCAB) outF[row * VOCAB + col] = v;
      }
    }
  }
}

// ---------------------------------------------------------------- attention
// one block per (n,h): S = QK^T/4 -> softmax -> O = P V, all via WMMA tiles
__global__ void k_attention(const _Float16* __restrict__ q16, const _Float16* __restrict__ k16,
                            const _Float16* __restrict__ vT16, _Float16* __restrict__ o16) {
  extern __shared__ char smem[];
  float*    sc  = (float*)smem;                        // [SP][SP] scores
  _Float16* p16 = (_Float16*)(smem + SP * SP * 4);     // [SP][KP] probabilities
  int nh = blockIdx.x; int n = nh >> 3; int h = nh & 7;
  int tid = threadIdx.x, wave = tid >> 5, lane = tid & 31;
  int mr = lane & 15, hi = lane >> 4;
  const _Float16* qb = q16 + (size_t)nh * SP * 16;
  const _Float16* kb = k16 + (size_t)nh * SP * 16;
  const _Float16* vb = vT16 + (size_t)nh * 16 * KP;

  if (wave < 7) {                       // 7 s-tiles of 16 rows
    int s0 = wave * 16;
    v8h z8 = {};
    AF a;
    a.h[0] = hi ? z8 : *(const v8h*)(qb + (size_t)(s0 + mr) * 16);      // d = 0..7 (hi=0)
    a.h[1] = hi ? z8 : *(const v8h*)(qb + (size_t)(s0 + mr) * 16 + 8);  // d = 8..15 (hi=0); K 16..31 zero pad
    for (int tt = 0; tt < 7; ++tt) {
      v16h b;
      if (hi == 0) {
        b = *(const v16h*)(kb + (size_t)(tt * 16 + mr) * 16);      // Bt[t][d], d=0..15
      } else {
        v16h zb = {}; b = zb;                                      // K 16..31 zero pad
      }
      v8f c = {};
      c = wmma_f16(a.v, b, c);
      int t = tt * 16 + mr;
#pragma unroll
      for (int i = 0; i < 8; i++) sc[(s0 + hi * 8 + i) * SP + t] = c[i] * 0.25f;
    }
  }
  __syncthreads();
  if (tid < SP) {                       // softmax over t < SEQ per row
    float* row = sc + tid * SP;
    float mx = -1e30f;
    for (int t = 0; t < SEQ; t++) mx = fmaxf(mx, row[t]);
    float ssum = 0.f;
    for (int t = 0; t < SEQ; t++) { float e = __expf(row[t] - mx); ssum += e; row[t] = e; }
    float inv = 1.f / ssum;
    _Float16* pr = p16 + tid * KP;
    for (int t = 0; t < SEQ; t++) pr[t] = (_Float16)(row[t] * inv);
    for (int t = SEQ; t < KP; t++) pr[t] = (_Float16)0.f;   // mask padded t
  }
  __syncthreads();
  if (wave < 7) {                       // O = P[16 x 128] @ V[128 x 16]
    int s0 = wave * 16;
    const _Float16* ar = p16 + (size_t)(s0 + mr) * KP;
    const _Float16* br = vb + (size_t)mr * KP;
    v8f c = {};
#pragma unroll
    for (int kc = 0; kc < 4; kc++) {
      AF a;
      a.v = *(const v16h*)(ar + kc * 32 + hi * 16);
      v16h b = *(const v16h*)(br + kc * 32 + hi * 16);
      c = wmma_f16(a.v, b, c);
    }
#pragma unroll
    for (int i = 0; i < 8; i++) {
      int s = s0 + hi * 8 + i;
      if (s < SEQ)
        o16[((size_t)s * N_NODES + n) * HID + h * 16 + mr] = (_Float16)c[i];
    }
  }
}

// ---------------------------------------------------------------- fused FF
// per block: 16 rows; A tile async-staged to LDS; mid = relu(A@W1+b1) stays in
// LDS (16x2048 f16); out = mid @ W2 + b2 + resid (f32).
__global__ void k_ff(const _Float16* __restrict__ tgt16, const float* __restrict__ tgtF,
                     const _Float16* __restrict__ w1T, const float* __restrict__ b1,
                     const _Float16* __restrict__ w2T, const float* __restrict__ b2,
                     float* __restrict__ outF) {
  extern __shared__ char smem[];
  _Float16* mid   = (_Float16*)smem;                      // [16][FFD] = 64KB
  _Float16* atile = (_Float16*)(smem + 16 * FFD * 2);     // [16][HID] = 4KB
  int m0 = blockIdx.x * 16;
  int tid = threadIdx.x, wave = tid >> 5, lane = tid & 31;
  int mr = lane & 15, hi = lane >> 4;

  // async-stage the 16x128 A tile (2048 halves, 8 per lane)
  async_ld16(tgt16 + (size_t)m0 * HID + tid * 8, atile + tid * 8);
  async_wait0();
  __syncthreads();

  const _Float16* arow = atile + (size_t)mr * HID;
  AF a[4];
#pragma unroll
  for (int kc = 0; kc < 4; kc++) {
    a[kc].v = *(const v16h*)(arow + kc * 32 + hi * 16);
  }
  for (int nt = wave; nt < FFD / 16; nt += 8) {
    int colb = nt * 16;
    const _Float16* brow = w1T + (size_t)(colb + mr) * HID;
    if (colb + 128 < FFD)
      __builtin_prefetch(w1T + (size_t)(colb + 128 + mr) * HID, 0, 0);
    v8f c = {};
#pragma unroll
    for (int kc = 0; kc < 4; kc++) {
      v16h b = *(const v16h*)(brow + kc * 32 + hi * 16);
      c = wmma_f16(a[kc].v, b, c);
    }
    float bv = b1[colb + mr];
#pragma unroll
    for (int i = 0; i < 8; i++) {
      float v = fmaxf(c[i] + bv, 0.f);
      mid[(size_t)(hi * 8 + i) * FFD + colb + mr] = (_Float16)v;
    }
  }
  __syncthreads();
  {
    int colb = wave * 16; int col = colb + mr;
    const _Float16* ar = mid + (size_t)mr * FFD;
    const _Float16* br = w2T + (size_t)col * FFD;
    v8f c = {};
    for (int k0 = 0; k0 < FFD; k0 += 32) {
      if (k0 + 256 < FFD) __builtin_prefetch(br + k0 + 256, 0, 0);
      AF a2;
      a2.v = *(const v16h*)(ar + k0 + hi * 16);
      v16h b = *(const v16h*)(br + k0 + hi * 16);
      c = wmma_f16(a2.v, b, c);
    }
    float bv = b2[col];
#pragma unroll
    for (int i = 0; i < 8; i++) {
      size_t row = (size_t)m0 + hi * 8 + i;
      outF[row * HID + col] = c[i] + bv + tgtF[row * HID + col];
    }
  }
}

// ---------------------------------------------------------------- layernorm
// one wave per 128-wide row; optional per-node additive vector (cross-attn)
__global__ void k_layernorm(const float* __restrict__ X, const float* __restrict__ addVec,
                            const float* __restrict__ w, const float* __restrict__ b,
                            float* __restrict__ outF, _Float16* __restrict__ outH, int M) {
  int row = blockIdx.x * 8 + (threadIdx.x >> 5);
  int lane = threadIdx.x & 31;
  if (row >= M) return;
  const float* x = X + (size_t)row * HID;
  const float* av = addVec ? addVec + (size_t)(row & (N_NODES - 1)) * HID : nullptr;
  float v[4];
#pragma unroll
  for (int j = 0; j < 4; j++) { int c = lane + j * 32; v[j] = x[c] + (av ? av[c] : 0.f); }
  float s = v[0] + v[1] + v[2] + v[3];
#pragma unroll
  for (int o = 16; o; o >>= 1) s += __shfl_xor(s, o);
  float mean = s * (1.f / HID);
  float q = 0.f;
#pragma unroll
  for (int j = 0; j < 4; j++) { float d = v[j] - mean; q += d * d; }
#pragma unroll
  for (int o = 16; o; o >>= 1) q += __shfl_xor(q, o);
  float rstd = rsqrtf(q * (1.f / HID) + 1e-5f);
#pragma unroll
  for (int j = 0; j < 4; j++) {
    int c = lane + j * 32;
    float y = (v[j] - mean) * rstd * w[c] + b[c];
    outF[(size_t)row * HID + c] = y;
    outH[(size_t)row * HID + c] = (_Float16)y;
  }
}

// ================================================================ host
extern "C" void kernel_launch(void* const* d_in, const int* in_sizes, int n_in,
                              void* d_out, int out_size, void* d_ws, size_t ws_size,
                              hipStream_t stream) {
  (void)in_sizes; (void)n_in; (void)out_size; (void)ws_size;
  const float* x      = (const float*)d_in[0];
  const int*   ei     = (const int*)d_in[1];
  const float* gcn1_w = (const float*)d_in[2];  const float* gcn1_b = (const float*)d_in[3];
  const float* gcn2_w = (const float*)d_in[4];  const float* gcn2_b = (const float*)d_in[5];
  const float* fc_w   = (const float*)d_in[6];  const float* fc_b   = (const float*)d_in[7];
  const float* sa_w   = (const float*)d_in[8];  const float* sa_b   = (const float*)d_in[9];
  const float* ca_w   = (const float*)d_in[10]; const float* ca_b   = (const float*)d_in[11];
  const float* ln_w   = (const float*)d_in[12]; const float* ln_b   = (const float*)d_in[13];
  const float* ff_w1  = (const float*)d_in[14]; const float* ff_b1  = (const float*)d_in[15];
  const float* ff_w2  = (const float*)d_in[16]; const float* ff_b2  = (const float*)d_in[17];
  const float* out_w  = (const float*)d_in[18]; const float* out_b  = (const float*)d_in[19];
  float* out = (float*)d_out;

  char* ws = (char*)d_ws; size_t off = 0;
  auto alloc = [&](size_t bytes) -> char* {
    char* p = ws + off; off = (off + bytes + 255) & ~(size_t)255; return p;
  };
  _Float16* gcn1T = (_Float16*)alloc((size_t)128 * 64 * 2);
  _Float16* gcn2T = (_Float16*)alloc((size_t)128 * 128 * 2);
  _Float16* fcT   = (_Float16*)alloc((size_t)128 * 128 * 2);
  _Float16* saT   = (_Float16*)alloc((size_t)LAYERS * 4 * 128 * 128 * 2);
  _Float16* caT   = (_Float16*)alloc((size_t)LAYERS * 4 * 128 * 128 * 2);
  _Float16* ff1T  = (_Float16*)alloc((size_t)LAYERS * FFD * HID * 2);
  _Float16* ff2T  = (_Float16*)alloc((size_t)LAYERS * HID * FFD * 2);
  _Float16* outT  = (_Float16*)alloc((size_t)64 * 128 * 2);
  float*    dinv  = (float*)alloc(N_NODES * 4);
  _Float16* qx16  = (_Float16*)alloc((size_t)N_NODES * NODE_DIM * 2);
  float*    lin   = (float*)alloc((size_t)N_NODES * HID * 4);
  float*    agg   = (float*)alloc((size_t)N_NODES * HID * 4);
  float*    hF    = (float*)alloc((size_t)N_NODES * HID * 4);
  _Float16* hH    = (_Float16*)alloc((size_t)N_NODES * HID * 2);
  float*    memF  = (float*)alloc((size_t)N_NODES * HID * 4);
  _Float16* memH  = (_Float16*)alloc((size_t)N_NODES * HID * 2);
  float*    vmF   = (float*)alloc((size_t)N_NODES * HID * 4);
  _Float16* vmH   = (_Float16*)alloc((size_t)N_NODES * HID * 2);
  float*    cvec  = (float*)alloc((size_t)N_NODES * HID * 4);
  float*    tgtF  = (float*)alloc((size_t)MTOK * HID * 4);
  _Float16* tgtH  = (_Float16*)alloc((size_t)MTOK * HID * 2);
  float*    residF= (float*)alloc((size_t)MTOK * HID * 4);
  _Float16* q16   = (_Float16*)alloc((size_t)N_NODES * HEADS * SP * 16 * 2);
  _Float16* k16   = (_Float16*)alloc((size_t)N_NODES * HEADS * SP * 16 * 2);
  _Float16* vT16  = (_Float16*)alloc((size_t)N_NODES * HEADS * 16 * KP * 2);
  _Float16* o16   = (_Float16*)alloc((size_t)MTOK * HID * 2);

  auto convW = [&](const float* W, _Float16* dst, int K, int N, int Nd) {
    int total = Nd * K;
    k_convT<<<(total + 255) / 256, 256, 0, stream>>>(W, dst, K, N, Nd);
  };
  // weights -> transposed f16
  convW(gcn1_w, gcn1T, 64, 128, 128);
  convW(gcn2_w, gcn2T, 128, 128, 128);
  convW(fc_w, fcT, 128, 128, 128);
  for (int l = 0; l < LAYERS; l++) {
    for (int j = 0; j < 4; j++) {
      convW(sa_w + (size_t)(l * 4 + j) * 128 * 128, saT + (size_t)(l * 4 + j) * 128 * 128, 128, 128, 128);
      convW(ca_w + (size_t)(l * 4 + j) * 128 * 128, caT + (size_t)(l * 4 + j) * 128 * 128, 128, 128, 128);
    }
    convW(ff_w1 + (size_t)l * HID * FFD, ff1T + (size_t)l * FFD * HID, HID, FFD, FFD);
    convW(ff_w2 + (size_t)l * FFD * HID, ff2T + (size_t)l * HID * FFD, FFD, HID, HID);
  }
  convW(out_w, outT, 128, VOCAB, 64);

  // quantum embedding -> f16
  {
    float c1 = cosf(0.05f) - sinf(0.05f), c2 = cosf(0.1f), c3 = sinf(0.1f);
    k_quantum<<<(N_NODES * NODE_DIM + 255) / 256, 256, 0, stream>>>(x, qx16, c1, c2, c3, N_NODES * NODE_DIM);
  }
  // degrees (with self loops) -> dinv
  k_deg_init<<<2, 256, 0, stream>>>(dinv);
  k_deg_count<<<EDGES / 256, 256, 0, stream>>>(ei + EDGES, dinv);
  k_deg_rsqrt<<<2, 256, 0, stream>>>(dinv);

  dim3 blk(256);
  auto gemm = [&](const _Float16* A, const _Float16* Bt, const float* bias,
                  float* oF, _Float16* oH, const float* resid,
                  int M, int K, int Ntiles, int flags, int mode) {
    dim3 grid((M + 127) / 128, Ntiles);
    k_gemm<<<grid, blk, 0, stream>>>(A, Bt, bias, oF, oH, resid, M, K, flags, mode);
  };

  // GCN layer 1
  gemm(qx16, gcn1T, nullptr, lin, nullptr, nullptr, N_NODES, NODE_DIM, 8, 0, 0);
  k_zero_f32<<<(N_NODES * HID + 255) / 256, 256, 0, stream>>>(agg, (size_t)N_NODES * HID);
  k_edge_agg<<<EDGES, 128, 0, stream>>>(ei, ei + EDGES, dinv, lin, agg);
  k_gcn_post<<<(N_NODES * HID + 255) / 256, 256, 0, stream>>>(lin, dinv, agg, gcn1_b, hF, hH);
  // GCN layer 2
  gemm(hH, gcn2T, nullptr, lin, nullptr, nullptr, N_NODES, HID, 8, 0, 0);
  k_zero_f32<<<(N_NODES * HID + 255) / 256, 256, 0, stream>>>(agg, (size_t)N_NODES * HID);
  k_edge_agg<<<EDGES, 128, 0, stream>>>(ei, ei + EDGES, dinv, lin, agg);
  k_gcn_post<<<(N_NODES * HID + 255) / 256, 256, 0, stream>>>(lin, dinv, agg, gcn2_b, hF, hH);
  // fc -> memory
  gemm(hH, fcT, fc_b, memF, memH, nullptr, N_NODES, HID, 8, 0, 0);

  // init tgt and padded attention buffers
  k_zero_f32<<<((MTOK * HID) + 255) / 256, 256, 0, stream>>>(tgtF, (size_t)MTOK * HID);
  k_zero_f16<<<((MTOK * HID) + 255) / 256, 256, 0, stream>>>(tgtH, (size_t)MTOK * HID);
  {
    size_t nq = (size_t)N_NODES * HEADS * SP * 16;
    size_t nv = (size_t)N_NODES * HEADS * 16 * KP;
    k_zero_f16<<<(nq + 255) / 256, 256, 0, stream>>>(q16, nq);
    k_zero_f16<<<(nq + 255) / 256, 256, 0, stream>>>(k16, nq);
    k_zero_f16<<<(nv + 255) / 256, 256, 0, stream>>>(vT16, nv);
  }

  int attn_shmem = SP * SP * 4 + SP * KP * 2;     // 78,848 B
  int ff_shmem   = 16 * FFD * 2 + 16 * HID * 2;   // 69,632 B

  for (int l = 0; l < LAYERS; l++) {
    const _Float16* sw = saT + (size_t)l * 4 * 128 * 128;
    const float*    sb = sa_b + (size_t)l * 4 * 128;
    // self-attention projections (packed f16 layouts)
    gemm(tgtH, sw + 0 * 16384, sb + 0,   nullptr, q16,  nullptr, MTOK, HID, 8, 0, 1);
    gemm(tgtH, sw + 1 * 16384, sb + 128, nullptr, k16,  nullptr, MTOK, HID, 8, 0, 1);
    gemm(tgtH, sw + 2 * 16384, sb + 256, nullptr, vT16, nullptr, MTOK, HID, 8, 0, 2);
    k_attention<<<N_NODES * HEADS, 256, attn_shmem, stream>>>(q16, k16, vT16, o16);
    // out projection + residual, then LN0
    gemm(o16, sw + 3 * 16384, sb + 384, residF, nullptr, tgtF, MTOK, HID, 8, 2, 0);
    k_layernorm<<<MTOK / 8, 256, 0, stream>>>(residF, nullptr,
        ln_w + (size_t)(l * 3 + 0) * HID, ln_b + (size_t)(l * 3 + 0) * HID, tgtF, tgtH, MTOK);
    // cross-attention: T==1 => softmax==1 => o = mem@Wv+bv, then @Wo+bo (per node)
    const _Float16* cw = caT + (size_t)l * 4 * 128 * 128;
    const float*    cb = ca_b + (size_t)l * 4 * 128;
    gemm(memH, cw + 2 * 16384, cb + 256, vmF, vmH, nullptr, N_NODES, HID, 8, 0, 0);
    gemm(vmH,  cw + 3 * 16384, cb + 384, cvec, nullptr, nullptr, N_NODES, HID, 8, 0, 0);
    k_layernorm<<<MTOK / 8, 256, 0, stream>>>(tgtF, cvec,
        ln_w + (size_t)(l * 3 + 1) * HID, ln_b + (size_t)(l * 3 + 1) * HID, tgtF, tgtH, MTOK);
    // fused feed-forward (2048-wide intermediate lives in LDS), then LN2
    k_ff<<<MTOK / 16, 256, ff_shmem, stream>>>(tgtH, tgtF,
        ff1T + (size_t)l * FFD * HID, ff_b1 + (size_t)l * FFD,
        ff2T + (size_t)l * HID * FFD, ff_b2 + (size_t)l * HID, residF);
    k_layernorm<<<MTOK / 8, 256, 0, stream>>>(residF, nullptr,
        ln_w + (size_t)(l * 3 + 2) * HID, ln_b + (size_t)(l * 3 + 2) * HID, tgtF, tgtH, MTOK);
  }

  // final vocab projection (N padded 50 -> 64)
  gemm(tgtH, outT, out_b, out, nullptr, nullptr, MTOK, HID, 4, 0, 3);
}
